// LatentAttention_38757784879512
// MI455X (gfx1250) — matrix operand order repaired and checked by reading, not compiled, not run on hardware
//
#include <hip/hip_runtime.h>

#define B_   8
#define S_   4096
#define E_   2048
#define L_   64
#define HS_  128
#define NEGV -1e20f

typedef __bf16 bf16;
typedef __bf16 v16bf __attribute__((ext_vector_type(16)));
typedef __bf16 v8bf  __attribute__((ext_vector_type(8)));
typedef float  v8f   __attribute__((ext_vector_type(8)));
typedef float  v4f   __attribute__((ext_vector_type(4)));
typedef unsigned int u32x4 __attribute__((ext_vector_type(4)));
typedef int          i32x4 __attribute__((ext_vector_type(4)));
typedef int          i32x8 __attribute__((ext_vector_type(8)));

#define LDS_ROW 72   // 144B padded row stride (128B data + 16B TDM pad)

__device__ __forceinline__ v8f v8f_zero() {
  v8f z;
#pragma unroll
  for (int i = 0; i < 8; ++i) z[i] = 0.0f;
  return z;
}

__device__ __forceinline__ v8f wmma_bf16(v16bf a, v16bf b, v8f c) {
  return __builtin_amdgcn_wmma_f32_16x16x32_bf16(
      false, a, false, b, (short)0, c, false, false);
}

__device__ __forceinline__ unsigned lds_off(const void* p) {
  // LDS aperture: addr[31:0] is the LDS byte offset
  return (unsigned)(unsigned long long)(uintptr_t)p;
}

// 2D bf16 tile -> LDS via Tensor Data Mover.  Rows of tile_d0 elements (128B
// here), row stride stride0 elements; LDS gets +16B pad per 128B row
// (pad_interval=4 -> 32 DWORDs, pad_amount=3 -> 4 DWORDs).
__device__ __forceinline__ void tdm_load_2d_bf16(
    unsigned lds_byte_addr, const void* gsrc,
    unsigned tile_d0, unsigned tile_d1,
    unsigned long long stride0_elems,
    unsigned tensor_d0, unsigned tensor_d1) {
  unsigned long long ga = (unsigned long long)(uintptr_t)gsrc;
  u32x4 g0;
  g0[0] = 1u;                                            // count=1, user D#
  g0[1] = lds_byte_addr;                                 // [63:32] lds_addr
  g0[2] = (unsigned)ga;                                  // global_addr lo
  g0[3] = (unsigned)((ga >> 32) & 0x1FFFFFFull) | (2u << 30);  // addr hi | type=2
  unsigned d0 = (1u << 16)                               // data_size = 2 bytes
              | (1u << 20) | (4u << 22) | (3u << 25);    // pad_en, every 128B, +16B
  i32x8 g1;
  g1[0] = (int)d0;
  g1[1] = (int)((tensor_d0 & 0xFFFFu) << 16);            // atomic_barrier_addr=0
  g1[2] = (int)((tensor_d0 >> 16) | ((tensor_d1 & 0xFFFFu) << 16));
  g1[3] = (int)((tensor_d1 >> 16) | ((tile_d0 & 0xFFFFu) << 16));
  g1[4] = (int)(tile_d1 & 0xFFFFu);                      // tile_dim2 = 0
  g1[5] = (int)(unsigned)(stride0_elems & 0xFFFFFFFFull);
  g1[6] = (int)(unsigned)((stride0_elems >> 32) & 0xFFFFull);  // dim1_stride=0
  g1[7] = 0;
  i32x4 z4 = {0, 0, 0, 0};
#if __clang_major__ >= 23
  i32x8 z8 = {0, 0, 0, 0, 0, 0, 0, 0};
  __builtin_amdgcn_tensor_load_to_lds(g0, g1, z4, z4, z8, 0);
#else
  __builtin_amdgcn_tensor_load_to_lds(g0, g1, z4, z4, 0);
#endif
}

// ---------------------------------------------------------------------------
// Kernel 1a: wproj[n][e] (bf16, K-contiguous).  n<64: W_dkv rows.
//            n>=64: absorbed_q[e][n-64] = sum_h Wq[h][e]*Wk[h][n-64].
// ---------------------------------------------------------------------------
__global__ void prep_wproj(const float* __restrict__ Wdkv,
                           const float* __restrict__ Wk,
                           const float* __restrict__ Wq,
                           bf16* __restrict__ wproj) {
  int idx = blockIdx.x * blockDim.x + threadIdx.x;
  int n = idx >> 11;
  int e = idx & (E_ - 1);
  float v;
  if (n < L_) {
    v = Wdkv[n * E_ + e];
  } else {
    int l = n - L_;
    float acc = 0.0f;
#pragma unroll 8
    for (int h = 0; h < HS_; ++h) acc += Wq[h * E_ + e] * Wk[h * L_ + l];
    v = acc;
  }
  wproj[n * E_ + e] = (bf16)v;
}

__global__ void prep_wv(const float* __restrict__ Wv, bf16* __restrict__ wv) {
  int i = blockIdx.x * blockDim.x + threadIdx.x;
  wv[i] = (bf16)Wv[i];
}

// ---------------------------------------------------------------------------
// Kernel 2: fused projection GEMM.  M=B*S, K=E, N=128 (64 latent | 64 q).
// x streamed from global; wproj K-chunks (128x64 bf16) TDM-staged into LDS,
// double buffered, one descriptor in flight behind compute.
// ---------------------------------------------------------------------------
__global__ __launch_bounds__(256)
void proj_kernel(const float* __restrict__ x, const bf16* __restrict__ wproj,
                 float* __restrict__ lat_f32, bf16* __restrict__ lat_bf16,
                 bf16* __restrict__ q_bf16) {
  __shared__ bf16 wlds[2][128][LDS_ROW];

  const int lane = threadIdx.x & 31;
  const int wave = threadIdx.x >> 5;
  const int l15  = lane & 15;
  const int half = lane >> 4;
  const int mbase = blockIdx.x * 128 + wave * 16;
  const bool w0 = (wave == 0);

  if (w0) {
    tdm_load_2d_bf16(lds_off(&wlds[0][0][0]), wproj,      64, 128, E_, E_, 128);
    tdm_load_2d_bf16(lds_off(&wlds[1][0][0]), wproj + 64, 64, 128, E_, E_, 128);
  }

  v8f c[8];
#pragma unroll
  for (int t = 0; t < 8; ++t) c[t] = v8f_zero();

  const float* xr = x + (size_t)(mbase + l15) * E_;

  for (int ck = 0; ck < E_ / 64; ++ck) {               // 32 K-chunks of 64
    if (w0) {
      if (ck < E_ / 64 - 1) __builtin_amdgcn_s_wait_tensorcnt(1);
      else                  __builtin_amdgcn_s_wait_tensorcnt(0);
    }
    __syncthreads();                                   // chunk ck resident

    const bf16* wb = &wlds[ck & 1][0][0];
    const int kb = ck * 64;
#pragma unroll
    for (int kk = 0; kk < 2; ++kk) {
      const int c0 = kb + kk * 32 + half * 16;
      v4f a0 = *(const v4f*)(xr + c0);
      v4f a1 = *(const v4f*)(xr + c0 + 4);
      v4f a2 = *(const v4f*)(xr + c0 + 8);
      v4f a3 = *(const v4f*)(xr + c0 + 12);
      v16bf a;
#pragma unroll
      for (int i = 0; i < 4; ++i) {
        a[i]      = (bf16)a0[i];
        a[4 + i]  = (bf16)a1[i];
        a[8 + i]  = (bf16)a2[i];
        a[12 + i] = (bf16)a3[i];
      }
      // hoist all 8 B-fragments before the WMMA chain (amortize DS latency)
      v16bf bfr[8];
#pragma unroll
      for (int t = 0; t < 8; ++t)
        bfr[t] = *(const v16bf*)(wb + (t * 16 + l15) * LDS_ROW + kk * 32 + half * 16);
#pragma unroll
      for (int t = 0; t < 8; ++t) c[t] = wmma_bf16(a, bfr[t], c[t]);
    }

    __syncthreads();                                   // all reads of buf done
    if (ck + 2 < E_ / 64 && w0)
      tdm_load_2d_bf16(lds_off(&wlds[ck & 1][0][0]), wproj + (size_t)(ck + 2) * 64,
                       64, 128, E_, E_, 128);
  }

  const float scale = 0.08838834764831845f;            // 128^-0.5, folded into q
#pragma unroll
  for (int t = 0; t < 8; ++t) {
#pragma unroll
    for (int r = 0; r < 8; ++r) {
      const int m = mbase + r + 8 * half;
      const int n = (t & 3) * 16 + l15;
      const float v = c[t][r];
      if (t < 4) {
        lat_f32[(size_t)m * L_ + n]  = v;
        lat_bf16[(size_t)m * L_ + n] = (bf16)v;
      } else {
        q_bf16[(size_t)m * L_ + n] = (bf16)(v * scale);
      }
    }
  }
}

// ---------------------------------------------------------------------------
// Kernel 3: v^T[b][h][s] = sum_l latent[b][s][l] * Wv[h][l]   (bf16 out)
// ---------------------------------------------------------------------------
__global__ __launch_bounds__(256)
void v_kernel(const bf16* __restrict__ lat_bf16, const bf16* __restrict__ wv,
              bf16* __restrict__ vT) {
  const int b = blockIdx.y;
  const int s = blockIdx.x * blockDim.x + threadIdx.x;
  const bf16* lr = lat_bf16 + ((size_t)b * S_ + s) * L_;
  float lv[L_];
#pragma unroll
  for (int l = 0; l < L_; ++l) lv[l] = (float)lr[l];
#pragma unroll 4
  for (int h = 0; h < HS_; ++h) {
    float acc = 0.0f;
#pragma unroll
    for (int l = 0; l < L_; ++l) acc += lv[l] * (float)wv[h * L_ + l];
    vT[((size_t)b * HS_ + h) * S_ + s] = (bf16)acc;
  }
}

// ---------------------------------------------------------------------------
// Kernel 4: causal flash attention.  WG = 128 threads (4 waves), 64 q-rows.
// kv tiles of 64: latent (64x64) and v^T (128x64) TDM-staged, double buffered.
// ---------------------------------------------------------------------------
__global__ __launch_bounds__(128)
void attn_kernel(const bf16* __restrict__ q_bf16,
                 const bf16* __restrict__ lat_bf16,
                 const bf16* __restrict__ vT,
                 float* __restrict__ out) {
  __shared__ bf16 klds[2][64][LDS_ROW];    // latent kv tile  [kv][l]
  __shared__ bf16 vlds[2][128][LDS_ROW];   // v^T tile        [h][kv]
  __shared__ bf16 pls[4][16][LDS_ROW];     // per-wave P slab

  const int lane = threadIdx.x & 31;
  const int wave = threadIdx.x >> 5;
  const int l15  = lane & 15;
  const int half = lane >> 4;
  const int b    = blockIdx.y;
  const int qb   = blockIdx.x * 64;
  const int jmax = blockIdx.x;
  const bool w0  = (wave == 0);

  const bf16* latB = lat_bf16 + (size_t)b * S_ * L_;
  const bf16* vB   = vT + (size_t)b * HS_ * S_;

  if (w0) {
    tdm_load_2d_bf16(lds_off(&klds[0][0][0]), latB, 64, 64, L_, L_, S_);
    tdm_load_2d_bf16(lds_off(&vlds[0][0][0]), vB,   64, 128, S_, S_, HS_);
    if (jmax >= 1) {
      tdm_load_2d_bf16(lds_off(&klds[1][0][0]), latB + 64ull * L_, 64, 64, L_, L_, S_);
      tdm_load_2d_bf16(lds_off(&vlds[1][0][0]), vB + 64,           64, 128, S_, S_, HS_);
    }
  }

  // Preload Q A-fragments (K = L = 64 -> 2 k-steps)
  const bf16* qrow = q_bf16 + ((size_t)b * S_ + qb + wave * 16 + l15) * L_;
  v16bf qa[2];
#pragma unroll
  for (int kk = 0; kk < 2; ++kk) {
    v8bf lo = *(const v8bf*)(qrow + kk * 32 + half * 16);
    v8bf hi = *(const v8bf*)(qrow + kk * 32 + half * 16 + 8);
#pragma unroll
    for (int i = 0; i < 8; ++i) { qa[kk][i] = lo[i]; qa[kk][8 + i] = hi[i]; }
  }

  v8f o[8];
#pragma unroll
  for (int t = 0; t < 8; ++t) o[t] = v8f_zero();
  float mrow[8], lrow[8];
#pragma unroll
  for (int r = 0; r < 8; ++r) { mrow[r] = -3.0e38f; lrow[r] = 0.0f; }

  for (int j = 0; j <= jmax; ++j) {
    if (w0) {
      if (j < jmax) __builtin_amdgcn_s_wait_tensorcnt(2);   // tile j resident
      else          __builtin_amdgcn_s_wait_tensorcnt(0);
    }
    __syncthreads();

    const bf16* kb16 = &klds[j & 1][0][0];
    const bf16* vb16 = &vlds[j & 1][0][0];
    const int kvb = j * 64;

    // ---- scores: 16 q rows x 64 kv (B-frags hoisted per k-step) ----
    v8f sc[4];
#pragma unroll
    for (int t = 0; t < 4; ++t) sc[t] = v8f_zero();
#pragma unroll
    for (int kk = 0; kk < 2; ++kk) {
      v16bf kf[4];
#pragma unroll
      for (int t = 0; t < 4; ++t)
        kf[t] = *(const v16bf*)(kb16 + (t * 16 + l15) * LDS_ROW + kk * 32 + half * 16);
#pragma unroll
      for (int t = 0; t < 4; ++t) sc[t] = wmma_bf16(qa[kk], kf[t], sc[t]);
    }

    // ---- causal mask on diagonal tile ----
    if (j == jmax) {
#pragma unroll
      for (int t = 0; t < 4; ++t) {
        const int k_g = kvb + t * 16 + l15;
#pragma unroll
        for (int r = 0; r < 8; ++r) {
          const int q_g = qb + wave * 16 + r + 8 * half;
          if (k_g > q_g) sc[t][r] = NEGV;
        }
      }
    }

    // ---- online softmax ----
#pragma unroll
    for (int r = 0; r < 8; ++r) {
      float v = fmaxf(fmaxf(sc[0][r], sc[1][r]), fmaxf(sc[2][r], sc[3][r]));
#pragma unroll
      for (int off = 1; off < 16; off <<= 1) v = fmaxf(v, __shfl_xor(v, off, 32));
      const float mnew = fmaxf(mrow[r], v);
      const float corr = __expf(mrow[r] - mnew);
      mrow[r] = mnew;
      float rs = 0.0f;
#pragma unroll
      for (int t = 0; t < 4; ++t) {
        const float p = __expf(sc[t][r] - mnew);
        sc[t][r] = p;
        rs += p;
      }
#pragma unroll
      for (int off = 1; off < 16; off <<= 1) rs += __shfl_xor(rs, off, 32);
      lrow[r] = lrow[r] * corr + rs;
#pragma unroll
      for (int t = 0; t < 8; ++t) o[t][r] *= corr;
    }

    // ---- P: C-layout -> A-layout via per-wave LDS slab ----
#pragma unroll
    for (int t = 0; t < 4; ++t)
#pragma unroll
      for (int r = 0; r < 8; ++r)
        pls[wave][r + 8 * half][t * 16 + l15] = (bf16)sc[t][r];

    v16bf pa[2];
#pragma unroll
    for (int kk = 0; kk < 2; ++kk) {
      const bf16* pr = &pls[wave][l15][0] + kk * 32 + half * 16;
      v8bf lo = *(const v8bf*)pr;
      v8bf hi = *(const v8bf*)(pr + 8);
#pragma unroll
      for (int i = 0; i < 8; ++i) { pa[kk][i] = lo[i]; pa[kk][8 + i] = hi[i]; }
    }

    // ---- O += P @ V (B-frags hoisted per k-step) ----
#pragma unroll
    for (int kk = 0; kk < 2; ++kk) {
      v16bf vf[8];
#pragma unroll
      for (int t = 0; t < 8; ++t)
        vf[t] = *(const v16bf*)(vb16 + (t * 16 + l15) * LDS_ROW + kk * 32 + half * 16);
#pragma unroll
      for (int t = 0; t < 8; ++t) o[t] = wmma_bf16(pa[kk], vf[t], o[t]);
    }

    __syncthreads();                                   // buffer reads complete
    if (w0 && j + 2 <= jmax) {
      const int kv2 = (j + 2) * 64;
      tdm_load_2d_bf16(lds_off(&klds[j & 1][0][0]), latB + (size_t)kv2 * L_, 64, 64, L_, L_, S_);
      tdm_load_2d_bf16(lds_off(&vlds[j & 1][0][0]), vB + kv2,                64, 128, S_, S_, HS_);
    }
  }

  // ---- epilogue: normalize and store ----
  float* ob = out + ((size_t)b * S_ + qb + wave * 16) * HS_;
#pragma unroll
  for (int t = 0; t < 8; ++t)
#pragma unroll
    for (int r = 0; r < 8; ++r) {
      const int m = r + 8 * half;
      ob[(size_t)m * HS_ + t * 16 + l15] = o[t][r] / lrow[r];
    }
}

// ---------------------------------------------------------------------------
extern "C" void kernel_launch(void* const* d_in, const int* in_sizes, int n_in,
                              void* d_out, int out_size, void* d_ws, size_t ws_size,
                              hipStream_t stream) {
  const float* x    = (const float*)d_in[0];
  const float* Wdkv = (const float*)d_in[1];
  const float* Wk   = (const float*)d_in[2];
  const float* Wv   = (const float*)d_in[3];
  const float* Wq   = (const float*)d_in[4];

  float* out     = (float*)d_out;                       // [B,S,128]
  float* lat_f32 = out + (size_t)B_ * S_ * HS_;         // [B,S,64]

  char* ws = (char*)d_ws;
  bf16* wproj = (bf16*)ws;                                        // 128*2048
  bf16* wv    = (bf16*)(ws + (size_t)128 * E_ * 2);               // 128*64
  bf16* qbuf  = (bf16*)(ws + (size_t)128 * E_ * 2 + 128 * 64 * 2);
  bf16* latb  = qbuf + (size_t)B_ * S_ * L_;
  bf16* vT    = latb + (size_t)B_ * S_ * L_;

  prep_wproj<<<(128 * E_) / 256, 256, 0, stream>>>(Wdkv, Wk, Wq, wproj);
  prep_wv<<<(128 * 64) / 256, 256, 0, stream>>>(Wv, wv);
  proj_kernel<<<(B_ * S_) / 128, 256, 0, stream>>>(x, wproj, lat_f32, latb, qbuf);
  v_kernel<<<dim3(S_ / 256, B_), 256, 0, stream>>>(latb, wv, vT);
  attn_kernel<<<dim3(S_ / 64, B_), 128, 0, stream>>>(qbuf, latb, vT, out);
}
